// CANN_39994735460546
// MI455X (gfx1250) — compile-verified, matrix operand order checked by reading, not run on hardware
//
#include <hip/hip_runtime.h>

typedef float v2f __attribute__((ext_vector_type(2)));
typedef float v8f __attribute__((ext_vector_type(8)));
typedef _Float16 v16h __attribute__((ext_vector_type(16)));

__constant__ float PS_POWERS[16] = {-5.0f, -4.0f, -3.0f, -2.0f, -1.5f, -1.0f, -0.5f, 0.0f,
                                    0.5f, 2.0f, (float)(1.0 / 3.0), 3.0f, 0.25f, 4.0f, 0.2f, 5.0f};
__constant__ float POLY_POWERS[5] = {2.0f, 5.0f, 8.0f, 11.0f, 14.0f};

// ---------------------------------------------------------------------------
// Stage 1: per-block partial sums of t and t^2 over all B*64 power-series
// values (deterministic tree reduction; no float atomics).
// ws layout (floats): [0]=mean [1]=1/std  [2 .. 2+2*nblk) = partials
// ---------------------------------------------------------------------------
__global__ void ps_partial_reduce(const float* __restrict__ pp,
                                  float* __restrict__ ws, int n_params) {
    __shared__ float sh_s[256];
    __shared__ float sh_q[256];
    const int tid = threadIdx.x;
    const int gid = blockIdx.x * 256 + tid;
    float s = 0.0f, q = 0.0f;
    if (gid < n_params) {
        const float x = pp[gid];
#pragma unroll
        for (int k = 0; k < 16; ++k) {
            const float v = powf(x, PS_POWERS[k]);
            s += v;
            q += v * v;
        }
    }
    sh_s[tid] = s;
    sh_q[tid] = q;
    __syncthreads();
    for (int off = 128; off > 0; off >>= 1) {
        if (tid < off) {
            sh_s[tid] += sh_s[tid + off];
            sh_q[tid] += sh_q[tid + off];
        }
        __syncthreads();
    }
    if (tid == 0) {
        ws[2 + 2 * blockIdx.x] = sh_s[0];
        ws[3 + 2 * blockIdx.x] = sh_q[0];
    }
}

// ---------------------------------------------------------------------------
// Stage 2: reduce partials -> mean and 1/std (unbiased, ddof=1)
// ---------------------------------------------------------------------------
__global__ void finalize_stats(float* __restrict__ ws, int nblocks, int ntotal) {
    __shared__ float sh_s[64];
    __shared__ float sh_q[64];
    const int tid = threadIdx.x;
    float s = 0.0f, q = 0.0f;
    for (int i = tid; i < nblocks; i += 64) {
        s += ws[2 + 2 * i];
        q += ws[3 + 2 * i];
    }
    sh_s[tid] = s;
    sh_q[tid] = q;
    __syncthreads();
    for (int off = 32; off > 0; off >>= 1) {
        if (tid < off) {
            sh_s[tid] += sh_s[tid + off];
            sh_q[tid] += sh_q[tid + off];
        }
        __syncthreads();
    }
    if (tid == 0) {
        const float N = (float)ntotal;
        const float mean = sh_s[0] / N;
        const float var = (sh_q[0] - sh_s[0] * sh_s[0] / N) / (N - 1.0f);
        ws[0] = mean;
        ws[1] = rsqrtf(var);
    }
}

// ---------------------------------------------------------------------------
// Stage 3: one wave (32 lanes) per 16-row tile.
//   features -> normalize -> ReLU -> LDS, then WMMA f32 16x16x4 for the
//   64->32 layer (N=32 as two 16-wide tiles, K=64 as 16 steps of 4),
//   then scalar 32->5 layer; store coefficients pre-scaled by POLY_POWERS.
// ---------------------------------------------------------------------------
__global__ __launch_bounds__(32) void coeff_kernel(
    const float* __restrict__ pp, const float* __restrict__ W1,
    const float* __restrict__ b1, const float* __restrict__ W2,
    const float* __restrict__ b2, const float* __restrict__ stats,
    float* __restrict__ sc) {
    __shared__ float tA[16][68];  // padded normalized-ReLU feature tile [16][64]
    __shared__ float tH[16][36];  // hidden layer tile [16][32]

    const int lane = threadIdx.x;
    const int rb = blockIdx.x * 16;
    const float mean = stats[0];
    const float istd = stats[1];

    for (int idx = lane; idx < 16 * 64; idx += 32) {
        const int r = idx >> 6;
        const int c = idx & 63;
        const float x = pp[(rb + r) * 4 + (c >> 4)];
        float v = powf(x, PS_POWERS[c & 15]);
        v = (v - mean) * istd;
        tA[r][c] = fmaxf(v, 0.0f);
    }
    __syncthreads();

    const int lr = lane & 15;
    const int hi = lane >> 4;

#if __has_builtin(__builtin_amdgcn_wmma_f32_16x16x4_f32)
    // Full-precision path: V_WMMA_F32_16X16X4_F32.
    // A 16x4 layout: lanes 0-15 hold K=k0,k0+1; lanes 16-31 hold K=k0+2,k0+3 (M=lane&15).
    // B 4x16 layout mirrors A with N across lanes. D: VGPR r -> rows r (lanes 0-15)
    // and r+8 (lanes 16-31), N = lane&15.
    for (int nb = 0; nb < 32; nb += 16) {
        v8f acc = {};
        const int n = nb + lr;
        for (int k0 = 0; k0 < 64; k0 += 4) {
            const int kk = k0 + 2 * hi;
            v2f a;
            a.x = tA[lr][kk];
            a.y = tA[lr][kk + 1];
            v2f b;
            b.x = W1[n * 64 + kk];
            b.y = W1[n * 64 + kk + 1];
            acc = __builtin_amdgcn_wmma_f32_16x16x4_f32(
                false, a, false, b, (short)0, acc, false, false);
        }
#pragma unroll
        for (int r = 0; r < 8; ++r) tH[r + 8 * hi][n] = acc[r] + b1[n];
    }
#else
    // Fallback: codegen-confirmed f16 WMMA, two K-steps of 32.
    // 16-bit A 16x32 layout: VGPR v (halves h): v<4 -> K = 2v+h (+8 for lanes 16-31),
    // v>=4 -> K = 16 + 2(v-4)+h (+8 for lanes 16-31); M = lane&15.
    for (int nb = 0; nb < 32; nb += 16) {
        v8f acc = {};
        const int n = nb + lr;
        for (int ks = 0; ks < 64; ks += 32) {
            v16h a, b;
#pragma unroll
            for (int e = 0; e < 16; ++e) {
                const int v = e >> 1;
                const int h = e & 1;
                const int k = ks + ((v < 4) ? (2 * v + h) : (16 + 2 * (v - 4) + h)) +
                              (hi ? 8 : 0);
                a[e] = (_Float16)tA[lr][k];
                b[e] = (_Float16)W1[n * 64 + k];
            }
            acc = __builtin_amdgcn_wmma_f32_16x16x32_f16(
                false, a, false, b, (short)0, acc, false, false);
        }
#pragma unroll
        for (int r = 0; r < 8; ++r) tH[r + 8 * hi][n] = acc[r] + b1[n];
    }
#endif
    __syncthreads();

    // Second linear layer (32 -> 5), pre-scale by POLY_POWERS[k].
    for (int idx = lane; idx < 80; idx += 32) {
        const int r = idx / 5;
        const int k = idx % 5;
        float acc = b2[k];
#pragma unroll
        for (int j = 0; j < 32; ++j) acc += tH[r][j] * W2[k * 32 + j];
        sc[(rb + r) * 5 + k] = acc * POLY_POWERS[k];
    }
}

// ---------------------------------------------------------------------------
// Stage 4: bandwidth-bound evaluation. float4 loads/stores (b128), uniform
// per-row coefficient loads (scalarized to s_load), e^3 power ladder + FMAs.
// ---------------------------------------------------------------------------
__device__ __forceinline__ float poly_eval(float e, float s0, float s1, float s2,
                                           float s3, float s4) {
    const float e2 = e * e;
    const float e3 = e2 * e;
    const float e4 = e3 * e;
    const float e7 = e4 * e3;
    const float e10 = e7 * e3;
    const float e13 = e10 * e3;
    return fmaf(s4, e13, fmaf(s3, e10, fmaf(s2, e7, fmaf(s1, e4, s0 * e))));
}

__global__ __launch_bounds__(256) void eta_poly_kernel(
    const float* __restrict__ eta, const float* __restrict__ sc,
    float* __restrict__ out, int L) {
    const int row = blockIdx.y;
    const long long base = (long long)row * L;
    const int j0 = (blockIdx.x * 256 + threadIdx.x) * 4;
    // Workgroup-uniform loads -> scalar registers broadcast across the wave.
    const float s0 = sc[row * 5 + 0];
    const float s1 = sc[row * 5 + 1];
    const float s2 = sc[row * 5 + 2];
    const float s3 = sc[row * 5 + 3];
    const float s4 = sc[row * 5 + 4];

    const float4 e = *(const float4*)(eta + base + j0);
    float4 r;
    r.x = poly_eval(e.x, s0, s1, s2, s3, s4);
    r.y = poly_eval(e.y, s0, s1, s2, s3, s4);
    r.z = poly_eval(e.z, s0, s1, s2, s3, s4);
    r.w = poly_eval(e.w, s0, s1, s2, s3, s4);
    *(float4*)(out + base + j0) = r;
}

extern "C" void kernel_launch(void* const* d_in, const int* in_sizes, int n_in,
                              void* d_out, int out_size, void* d_ws, size_t ws_size,
                              hipStream_t stream) {
    const float* pp = (const float*)d_in[0];   // [B,4]
    const float* eta = (const float*)d_in[1];  // [B,L]
    const float* W1 = (const float*)d_in[2];   // [32,64]
    const float* b1 = (const float*)d_in[3];   // [32]
    const float* W2 = (const float*)d_in[4];   // [5,32]
    const float* b2 = (const float*)d_in[5];   // [5]
    float* out = (float*)d_out;

    const int B = in_sizes[0] / 4;
    const int L = in_sizes[1] / B;

    float* wsf = (float*)d_ws;        // [0]=mean [1]=1/std [2..] partials
    float* sc = wsf + 256;            // scaled coefficients [B,5]

    const int n_params = B * 4;
    const int nblk = (n_params + 255) / 256;

    ps_partial_reduce<<<nblk, 256, 0, stream>>>(pp, wsf, n_params);
    finalize_stats<<<1, 64, 0, stream>>>(wsf, nblk, B * 64);
    coeff_kernel<<<B / 16, 32, 0, stream>>>(pp, W1, b1, W2, b2, wsf, sc);

    dim3 grid(L / (256 * 4), B);
    eta_poly_kernel<<<grid, 256, 0, stream>>>(eta, sc, out, L);
}
